// GAU_Attention_61495341744294
// MI455X (gfx1250) — matrix operand order repaired and checked by reading, not compiled, not run on hardware
//
#include <hip/hip_runtime.h>
#include <math.h>

typedef __attribute__((ext_vector_type(16))) __bf16 v16bf;
typedef __attribute__((ext_vector_type(8)))  __bf16 v8bf;
typedef __attribute__((ext_vector_type(8)))  float  v8f;
typedef __attribute__((ext_vector_type(4)))  int    i32x4;

#define BM 128
#define BN 128
#define BKS 64          // K-chunk per LDS stage = two 16x16x32 WMMA substeps

// Epilogue modes
#define EP_SILU_BIAS 0
#define EP_RELU2     1
#define EP_MUL       2
#define EP_BIAS_F32  3

// ---- CDNA5 async global->LDS copy (direct, no VGPR round trip) ------------
#if defined(__has_builtin)
# if __has_builtin(__builtin_amdgcn_global_load_async_to_lds_b128)
#  define ASYNC_COPY 1
# endif
#endif
#ifndef ASYNC_COPY
# define ASYNC_COPY 0
#endif

#if ASYNC_COPY && defined(__has_builtin) && __has_builtin(__builtin_amdgcn_s_wait_asynccnt)
# define WAIT_ASYNC() __builtin_amdgcn_s_wait_asynccnt(0)
#elif ASYNC_COPY
# define WAIT_ASYNC() asm volatile("s_wait_asynccnt 0x0" ::: "memory")
#else
# define WAIT_ASYNC()
#endif

__device__ __forceinline__ void cp16(__bf16* l, const __bf16* g)
{
#if ASYNC_COPY
    // global_load_async_to_lds_b128: (v4i AS1* gaddr, v4i AS3* lds, off, cpol)
    __builtin_amdgcn_global_load_async_to_lds_b128(
        (__attribute__((address_space(1))) i32x4*)g,
        (__attribute__((address_space(3))) i32x4*)l, 0, 0);
#else
    *(uint4*)l = *(const uint4*)g;     // global_load_b128 + ds_store_b128
#endif
}

// ---------------------------------------------------------------------------
// LayerNorm (f32 in) -> bf16 out.  One block per row of H elements.
// ---------------------------------------------------------------------------
__global__ __launch_bounds__(256)
void ln_to_bf16(const float* __restrict__ x, const float* __restrict__ gamma,
                const float* __restrict__ beta, __bf16* __restrict__ out, int Hdim)
{
    __shared__ float rs[256];
    __shared__ float rq[256];
    const size_t row = blockIdx.x;
    const float* xr = x + row * (size_t)Hdim;
    float s = 0.f, q = 0.f;
    for (int i = threadIdx.x; i < Hdim; i += 256) {
        float v = xr[i];
        s += v; q += v * v;
    }
    rs[threadIdx.x] = s; rq[threadIdx.x] = q;
    __syncthreads();
    for (int off = 128; off > 0; off >>= 1) {
        if ((int)threadIdx.x < off) {
            rs[threadIdx.x] += rs[threadIdx.x + off];
            rq[threadIdx.x] += rq[threadIdx.x + off];
        }
        __syncthreads();
    }
    const float mean = rs[0] / (float)Hdim;
    const float var  = rq[0] / (float)Hdim - mean * mean;
    const float inv  = rsqrtf(var + 1e-5f);
    __bf16* orow = out + row * (size_t)Hdim;
    for (int i = threadIdx.x; i < Hdim; i += 256)
        orow[i] = (__bf16)((xr[i] - mean) * inv * gamma[i] + beta[i]);
}

__global__ __launch_bounds__(256)
void f32_to_bf16(const float* __restrict__ in, __bf16* __restrict__ out, int n)
{
    int i = blockIdx.x * 256 + threadIdx.x;
    if (i < n) out[i] = (__bf16)in[i];
}

// ---------------------------------------------------------------------------
// bf16 GEMM, C = epilogue(A * Bt^T).
//   A  : (M x K) row-major bf16 (+ b*strideA per batch)
//   Bt : (N x K) row-major bf16 (K-contiguous; weights / V^T are native)
// Block tile 128x128; 8 waves, each owns a 64x32 C sub-tile (4x2 WMMA tiles).
// Double-buffered LDS, BKS=64 per stage (2 WMMA K-substeps), async copies
// (GLOBAL_LOAD_ASYNC_TO_LDS_B128 + s_wait_asynccnt) when available.
// ---------------------------------------------------------------------------
template<int MODE, bool TRANS_OUT>
__global__ __launch_bounds__(256)
void gemm_bf16_wmma(const __bf16* __restrict__ A, int lda, long strideA,
                    const __bf16* __restrict__ Bt, int ldb, long strideB,
                    void* __restrict__ Cout, int ldc, long strideC,
                    const float* __restrict__ bias,
                    const __bf16* __restrict__ mul, long strideMul,
                    float scale, int K)
{
    __shared__ __bf16 lA[2][BM * BKS];   // 2 x 16 KB
    __shared__ __bf16 lB[2][BN * BKS];   // 2 x 16 KB

    const int tid  = threadIdx.x;
    const int lane = tid & 31;
    const int wave = tid >> 5;
    const int wm   = wave & 1;           // 2 waves along M
    const int wn   = wave >> 1;          // 4 waves along N
    const int bat  = blockIdx.z;

    const __bf16* Ab = A  + (size_t)bat * (size_t)strideA;
    const __bf16* Bb = Bt + (size_t)bat * (size_t)strideB;

    const int tile_m = blockIdx.y * BM;
    const int tile_n = blockIdx.x * BN;

    const v8f vzero = {0.f,0.f,0.f,0.f,0.f,0.f,0.f,0.f};
    v8f acc[4][2];
    #pragma unroll
    for (int i = 0; i < 4; i++)
        #pragma unroll
        for (int j = 0; j < 2; j++) acc[i][j] = vzero;

    // Staging: tile = 128 rows x 64 cols bf16 = 1024 x 16B chunks; 4/thread.
    const int srow = tid >> 3;           // 0..31 (+32*i)
    const int scol = (tid & 7) * 8;      // element offset 0..56

    auto stage = [&](int k0, int buf) {
        #pragma unroll
        for (int i = 0; i < 4; i++) {
            const int r = srow + i * 32;
            cp16(&lA[buf][r * BKS + scol],
                 &Ab[(size_t)(tile_m + r) * lda + k0 + scol]);
            cp16(&lB[buf][r * BKS + scol],
                 &Bb[(size_t)(tile_n + r) * ldb + k0 + scol]);
        }
    };

    // WMMA fragment addressing (A and B both use contiguous k = 16*(lane/16) + s)
    const int klo_a = (lane >> 4) * 16;  // A: 16B chunks at klo_a, klo_a+8
    const int klo_b = (lane >> 4) * 16;  // B: 16 contiguous K at klo_b
    const int l15   = lane & 15;

    const int nk = K / BKS;
    stage(0, 0);
    WAIT_ASYNC();
    __syncthreads();

    for (int ks = 0; ks < nk; ks++) {
        const int cur = ks & 1;
        if (ks + 1 < nk) stage((ks + 1) * BKS, cur ^ 1);  // overlap with WMMA

        const __bf16* A_l = lA[cur];
        const __bf16* B_l = lB[cur];
        #pragma unroll
        for (int s = 0; s < 2; s++) {            // two 16x16x32 K-substeps
            const int kb = s * 32;
            v16bf afrag[4];
            #pragma unroll
            for (int mi = 0; mi < 4; mi++) {
                const int row = wm * 64 + mi * 16 + l15;
                v8bf lo = *(const v8bf*)&A_l[row * BKS + kb + klo_a];
                v8bf hi = *(const v8bf*)&A_l[row * BKS + kb + klo_a + 8];
                afrag[mi] = __builtin_shufflevector(lo, hi,
                    0,1,2,3,4,5,6,7,8,9,10,11,12,13,14,15);
            }
            v16bf bfrag[2];
            #pragma unroll
            for (int ni = 0; ni < 2; ni++) {
                const int col = wn * 32 + ni * 16 + l15;
                bfrag[ni] = *(const v16bf*)&B_l[col * BKS + kb + klo_b];
            }
            #pragma unroll
            for (int mi = 0; mi < 4; mi++)
                #pragma unroll
                for (int ni = 0; ni < 2; ni++)
                    acc[mi][ni] = __builtin_amdgcn_wmma_f32_16x16x32_bf16(
                        false, afrag[mi], false, bfrag[ni],
                        (short)0, acc[mi][ni], false, false);
        }
        WAIT_ASYNC();        // next tile fully landed in LDS
        __syncthreads();     // dscnt-fenced barrier: reads done, writes visible
    }

    // ---- epilogue + store (C: VGPR r -> row m0 + 8*(lane>=16) + r) ----
    #pragma unroll
    for (int mi = 0; mi < 4; mi++) {
        const int m0 = tile_m + wm * 64 + mi * 16 + ((lane >> 4) << 3);
        #pragma unroll
        for (int ni = 0; ni < 2; ni++) {
            const int n = tile_n + wn * 32 + ni * 16 + l15;
            v8f c = acc[mi][ni];
            #pragma unroll
            for (int r = 0; r < 8; r++) {
                const int row = m0 + r;
                float v = c[r];
                if (MODE == EP_SILU_BIAS) {
                    v += bias[n];
                    v = v / (1.f + __expf(-v));      // silu
                    const __bf16 o = (__bf16)v;
                    if (TRANS_OUT) {
                        const int bb = row >> 11;    // T == 2048
                        const int t  = row & 2047;
                        ((__bf16*)Cout)[(size_t)bb * (size_t)strideC +
                                        (size_t)n * ldc + t] = o;
                    } else {
                        ((__bf16*)Cout)[(size_t)bat * (size_t)strideC +
                                        (size_t)row * ldc + n] = o;
                    }
                } else if (MODE == EP_RELU2) {
                    v *= scale;
                    v  = fmaxf(v, 0.f);
                    v  = v * v;                       // relu^2 attention
                    ((__bf16*)Cout)[(size_t)bat * (size_t)strideC +
                                    (size_t)row * ldc + n] = (__bf16)v;
                } else if (MODE == EP_MUL) {
                    const float u = (float)mul[(size_t)bat * (size_t)strideMul +
                                               (size_t)row * ldc + n];
                    ((__bf16*)Cout)[(size_t)bat * (size_t)strideC +
                                    (size_t)row * ldc + n] = (__bf16)(v * u);
                } else { // EP_BIAS_F32
                    v += bias[n];
                    ((float*)Cout)[(size_t)bat * (size_t)strideC +
                                   (size_t)row * ldc + n] = v;
                }
            }
        }
    }
}

// ---------------------------------------------------------------------------
extern "C" void kernel_launch(void* const* d_in, const int* in_sizes, int n_in,
                              void* d_out, int out_size, void* d_ws, size_t ws_size,
                              hipStream_t stream)
{
    const float* hs   = (const float*)d_in[0];
    const float* lng  = (const float*)d_in[1];
    const float* lnb  = (const float*)d_in[2];
    const float* Wu_w = (const float*)d_in[3];
    const float* Wu_b = (const float*)d_in[4];
    const float* Wq_w = (const float*)d_in[5];
    const float* Wq_b = (const float*)d_in[6];
    const float* Wk_w = (const float*)d_in[7];
    const float* Wk_b = (const float*)d_in[8];
    const float* Wv_w = (const float*)d_in[9];
    const float* Wv_b = (const float*)d_in[10];
    const float* Wo_w = (const float*)d_in[11];
    const float* Wo_b = (const float*)d_in[12];
    float* out = (float*)d_out;

    const int Bn = 4, T = 2048, Hd = 768;
    const int BT = Bn * T;
    const size_t WSZ = (size_t)Hd * Hd;

    char* p = (char*)d_ws;
    auto alloc_bf = [&](size_t nelem) {
        __bf16* r = (__bf16*)p;
        p += ((nelem * 2 + 255) & ~(size_t)255);
        return r;
    };
    __bf16* wub = alloc_bf(WSZ);
    __bf16* wqb = alloc_bf(WSZ);
    __bf16* wkb = alloc_bf(WSZ);
    __bf16* wvb = alloc_bf(WSZ);
    __bf16* wob = alloc_bf(WSZ);
    __bf16* xb  = alloc_bf((size_t)BT * Hd);
    __bf16* ub  = alloc_bf((size_t)BT * Hd);
    __bf16* qb  = alloc_bf((size_t)BT * Hd);
    __bf16* kb  = alloc_bf((size_t)BT * Hd);
    __bf16* vt  = alloc_bf((size_t)Bn * Hd * T);   // V^T (B, D, T)
    __bf16* sb  = alloc_bf((size_t)Bn * T * T);    // relu^2 scores
    __bf16* gb  = alloc_bf((size_t)BT * Hd);       // U * (A@V)

    const int cgrid = (int)((WSZ + 255) / 256);
    f32_to_bf16<<<cgrid, 256, 0, stream>>>(Wu_w, wub, (int)WSZ);
    f32_to_bf16<<<cgrid, 256, 0, stream>>>(Wq_w, wqb, (int)WSZ);
    f32_to_bf16<<<cgrid, 256, 0, stream>>>(Wk_w, wkb, (int)WSZ);
    f32_to_bf16<<<cgrid, 256, 0, stream>>>(Wv_w, wvb, (int)WSZ);
    f32_to_bf16<<<cgrid, 256, 0, stream>>>(Wo_w, wob, (int)WSZ);

    ln_to_bf16<<<BT, 256, 0, stream>>>(hs, lng, lnb, xb, Hd);

    const dim3 blk(256);
    const dim3 gProj(Hd / BN, BT / BM, 1);     // (6, 64)
    const long sTH = (long)T * Hd;

    gemm_bf16_wmma<EP_SILU_BIAS, false><<<gProj, blk, 0, stream>>>(
        xb, Hd, 0, wub, Hd, 0, ub, Hd, 0, Wu_b, nullptr, 0, 1.f, Hd);
    gemm_bf16_wmma<EP_SILU_BIAS, false><<<gProj, blk, 0, stream>>>(
        xb, Hd, 0, wqb, Hd, 0, qb, Hd, 0, Wq_b, nullptr, 0, 1.f, Hd);
    gemm_bf16_wmma<EP_SILU_BIAS, false><<<gProj, blk, 0, stream>>>(
        xb, Hd, 0, wkb, Hd, 0, kb, Hd, 0, Wk_b, nullptr, 0, 1.f, Hd);
    gemm_bf16_wmma<EP_SILU_BIAS, true><<<gProj, blk, 0, stream>>>(
        xb, Hd, 0, wvb, Hd, 0, vt, T, (long)Hd * T, Wv_b, nullptr, 0, 1.f, Hd);

    const float inv_sqrt_T = 0.022097086912079608f;   // 1/sqrt(2048)
    const dim3 gScore(T / BN, T / BM, Bn);            // (16, 16, 4)
    gemm_bf16_wmma<EP_RELU2, false><<<gScore, blk, 0, stream>>>(
        qb, Hd, sTH, kb, Hd, sTH, sb, T, (long)T * T,
        nullptr, nullptr, 0, inv_sqrt_T, Hd);

    const dim3 gAV(Hd / BN, T / BM, Bn);              // (6, 16, 4)
    gemm_bf16_wmma<EP_MUL, false><<<gAV, blk, 0, stream>>>(
        sb, T, (long)T * T, vt, T, (long)Hd * T, gb, Hd, sTH,
        nullptr, ub, sTH, 1.f, T);

    gemm_bf16_wmma<EP_BIAS_F32, false><<<gProj, blk, 0, stream>>>(
        gb, Hd, 0, wob, Hd, 0, out, Hd, 0, Wo_b, nullptr, 0, 1.f, Hd);

    (void)in_sizes; (void)n_in; (void)out_size; (void)ws_size;
}
